// Generator_37915971289383
// MI455X (gfx1250) — matrix operand order repaired and checked by reading, not compiled, not run on hardware
//
#include <hip/hip_runtime.h>
#include <cmath>

typedef __attribute__((ext_vector_type(16))) _Float16 v16h;
typedef __attribute__((ext_vector_type(8)))  float    v8f;

#define SQRT2F 1.4142135623730951f

__device__ __forceinline__ float lrelu(float x) { return x > 0.f ? x : 0.2f * x; }

// WMMA-B swizzled weight offset: lane-contiguous 16-half groups.
// k = ci*9+tap, layout: [k>>5][cout][(k>>4)&1][k&15]
__device__ __forceinline__ size_t swz(int k, int co, int Cout) {
  return ((((size_t)(k >> 5) * (size_t)Cout + (size_t)co) * 2 + ((k >> 4) & 1)) << 4) + (k & 15);
}

// ---------------------------------------------------------------------------
// Mapping network: normalize styles, 4x equal_linear(lr_mul=0.01, act) -> z
// ---------------------------------------------------------------------------
__global__ __launch_bounds__(512)
void k_mapping(const float* __restrict__ styles,
               const float* __restrict__ W0, const float* __restrict__ b0,
               const float* __restrict__ W1, const float* __restrict__ b1,
               const float* __restrict__ W2, const float* __restrict__ b2,
               const float* __restrict__ W3, const float* __restrict__ b3,
               float* __restrict__ zout) {
  __shared__ float zs[4][512];
  __shared__ float nxt[4][512];
  __shared__ float red[512];
  int t = threadIdx.x;
  for (int b = 0; b < 4; ++b) zs[b][t] = styles[b * 512 + t];
  __syncthreads();
  for (int b = 0; b < 4; ++b) {
    red[t] = zs[b][t] * zs[b][t];
    __syncthreads();
    for (int off = 256; off > 0; off >>= 1) {
      if (t < off) red[t] += red[t + off];
      __syncthreads();
    }
    float r = rsqrtf(red[0] * (1.f / 512.f) + 1e-8f);
    __syncthreads();
    zs[b][t] *= r;
  }
  __syncthreads();
  const float scale = 0.01f / 22.627416997969522f;  // lr_mul/sqrt(512)
  const float* Ws[4] = {W0, W1, W2, W3};
  const float* bs[4] = {b0, b1, b2, b3};
  for (int L = 0; L < 4; ++L) {
    const float* W = Ws[L];
    const float* bb = bs[L];
    float brow = bb[t] * 0.01f;
    for (int b = 0; b < 4; ++b) {
      float acc = 0.f;
      for (int i = 0; i < 512; ++i) acc += zs[b][i] * W[t * 512 + i];
      acc = acc * scale + brow;
      nxt[b][t] = lrelu(acc) * SQRT2F;
    }
    __syncthreads();
    for (int b = 0; b < 4; ++b) zs[b][t] = nxt[b][t];
    __syncthreads();
  }
  for (int b = 0; b < 4; ++b) zout[b * 512 + t] = zs[b][t];
}

// ---------------------------------------------------------------------------
// style = z @ (sW/sqrt(512)).T + sb
// ---------------------------------------------------------------------------
__global__ void k_style(const float* __restrict__ z, const float* __restrict__ sW,
                        const float* __restrict__ sb, float* __restrict__ style, int Cin) {
  int idx = blockIdx.x * blockDim.x + threadIdx.x;
  if (idx >= 4 * Cin) return;
  int b = idx / Cin, c = idx - b * Cin;
  const float scale = 0.044194173824159216f;  // 1/sqrt(512)
  float s = 0.f;
  const float* zr = z + b * 512;
  const float* wr = sW + (size_t)c * 512;
  for (int i = 0; i < 512; ++i) s += zr[i] * wr[i];
  style[idx] = s * scale + sb[c];
}

// ---------------------------------------------------------------------------
// demod[b][co] = rsqrt(sum_{ci,t} (mscale*W*style)^2 + 1e-8)
// ---------------------------------------------------------------------------
__global__ __launch_bounds__(256)
void k_demod(const float* __restrict__ W, const float* __restrict__ style,
             float* __restrict__ demod, int Cin, int Cout, float mscale) {
  int bc = blockIdx.x;
  int b = bc / Cout, co = bc - b * Cout;
  int K = Cin * 9;
  float s = 0.f;
  for (int j = threadIdx.x; j < K; j += 256) {
    int ci = j / 9;
    float wv = mscale * W[(size_t)co * K + j] * style[b * Cin + ci];
    s += wv * wv;
  }
  __shared__ float red[256];
  red[threadIdx.x] = s;
  __syncthreads();
  for (int off = 128; off > 0; off >>= 1) {
    if (threadIdx.x < off) red[threadIdx.x] += red[threadIdx.x + off];
    __syncthreads();
  }
  if (threadIdx.x == 0) demod[bc] = rsqrtf(red[0] + 1e-8f);
}

// ---------------------------------------------------------------------------
// Non-up modulated weights -> f16, swizzled [b][swz(k,co)]
// ---------------------------------------------------------------------------
__global__ void k_prepw(const float* __restrict__ W, const float* __restrict__ style,
                        const float* __restrict__ demod, _Float16* __restrict__ out,
                        int Cin, int Cout, float mscale) {
  int K = Cin * 9;
  long long total = 4LL * K * Cout;
  long long idx = (long long)blockIdx.x * blockDim.x + threadIdx.x;
  if (idx >= total) return;
  int co = (int)(idx % Cout);
  int k = (int)((idx / Cout) % K);
  int b = (int)(idx / ((long long)Cout * K));
  int ci = k / 9;
  int t = k - ci * 9;
  float wv = W[((size_t)co * Cin + ci) * 9 + t] * mscale * style[b * Cin + ci] *
             demod[b * Cout + co];
  out[(size_t)b * K * Cout + swz(k, co, Cout)] = (_Float16)wv;
}

// ---------------------------------------------------------------------------
// Up layers: fold conv_transpose(stride2,3x3)+blur4 into 4 per-parity 3x3
// kernels: Weff[py][px][ry][rx] =
//   (1/16) * sum_{a,b} w[2-a][2-b] * k1[2ry+(1-py)-a] * k1[2rx+(1-px)-b]
// ---------------------------------------------------------------------------
__global__ void k_prepw_up(const float* __restrict__ W, const float* __restrict__ style,
                           const float* __restrict__ demod, _Float16* __restrict__ out,
                           int Cin, int Cout, float mscale) {
  long long total = 4LL * Cout * Cin;
  long long idx = (long long)blockIdx.x * blockDim.x + threadIdx.x;
  if (idx >= total) return;
  int ci = (int)(idx % Cin);
  int co = (int)((idx / Cin) % Cout);
  int b = (int)(idx / ((long long)Cin * Cout));
  float sf = mscale * style[b * Cin + ci] * demod[b * Cout + co];
  float wm[3][3];
  for (int ky = 0; ky < 3; ++ky)
    for (int kx = 0; kx < 3; ++kx)
      wm[ky][kx] = W[((size_t)co * Cin + ci) * 9 + ky * 3 + kx] * sf;
  const float k1[4] = {1.f, 3.f, 3.f, 1.f};
  size_t K = (size_t)Cin * 9;
  for (int py = 0; py < 2; ++py)
    for (int px = 0; px < 2; ++px) {
      int cls = py * 2 + px;
      _Float16* ob = out + ((size_t)(b * 4 + cls)) * K * Cout;
      for (int ry = 0; ry < 3; ++ry)
        for (int rx = 0; rx < 3; ++rx) {
          int sy = 2 * ry + (1 - py);
          int sx = 2 * rx + (1 - px);
          float acc = 0.f;
          for (int a = 0; a < 3; ++a) {
            int dy = sy - a;
            if (dy < 0 || dy > 3) continue;
            for (int bb = 0; bb < 3; ++bb) {
              int dx = sx - bb;
              if (dx < 0 || dx > 3) continue;
              acc += wm[2 - a][2 - bb] * k1[dy] * k1[dx];
            }
          }
          acc *= (1.f / 16.f);
          ob[swz(ci * 9 + ry * 3 + rx, co, Cout)] = (_Float16)acc;
        }
    }
}

// plain (unmodulated) weight -> swizzled f16 (cond conv2)
__global__ void k_prep_plainw(const float* __restrict__ W, _Float16* __restrict__ out,
                              int Cin, int Cout) {
  long long total = (long long)Cout * Cin * 9;
  long long idx = (long long)blockIdx.x * blockDim.x + threadIdx.x;
  if (idx >= total) return;
  int t = (int)(idx % 9);
  int ci = (int)((idx / 9) % Cin);
  int co = (int)(idx / (9LL * Cin));
  out[swz(ci * 9 + t, co, Cout)] = (_Float16)W[idx];
}

// ---------------------------------------------------------------------------
// Implicit-GEMM 3x3 conv via V_WMMA_F32_16X16X32_F16.
// Block = 4 waves sharing one 16-pixel M tile (A staged in LDS in WMMA lane
// layout, double-buffered); each wave owns 32 couts (2 N-tiles, 2 acc chains).
// K unrolled by 64 -> 4 static v_wmma per iteration.
// up=1: per-parity-class gather conv (transposed-conv+blur folded weights).
// ---------------------------------------------------------------------------
__global__ __launch_bounds__(128)
void k_conv_wmma(const _Float16* __restrict__ x, const _Float16* __restrict__ w,
                 size_t wzStride, _Float16* __restrict__ out,
                 const float* __restrict__ noise, const float* __restrict__ nwp,
                 const float* __restrict__ ab, int Cin, int Cout, int H, int W,
                 int Ho, int Wo, int stride, int upMode, float actScale) {
  __shared__ _Float16 ldsA[2][512];  // 16 rows x 32 K halves, WMMA lane layout

  int zi = blockIdx.z;
  int b, py = 0, px = 0;
  if (upMode) { b = zi >> 2; py = (zi >> 1) & 1; px = zi & 1; }
  else b = zi;

  int Hq = upMode ? (Ho >> 1) : Ho;
  int Wq = upMode ? (Wo >> 1) : Wo;
  int K = Cin * 9;
  int m0 = blockIdx.x * 16;
  int n0 = (blockIdx.y * 4 + threadIdx.y) * 32;
  int lane = threadIdx.x;
  int tid = threadIdx.y * 32 + lane;
  int n = lane & 15;
  int gb = lane >> 4;

  const _Float16* xb = x + (size_t)b * Cin * H * W;
  const _Float16* wb = w + (size_t)zi * wzStride;
  const _Float16* wl0 = wb + ((((size_t)(n0 + n)) * 2 + (size_t)gb) << 4);
  const _Float16* wl1 = wb + ((((size_t)(n0 + 16 + n)) * 2 + (size_t)gb) << 4);
  size_t wchunk = (size_t)Cout << 5;  // halves per 32-K chunk

  // Staging constants: thread owns 4 consecutive LDS slots s0..s0+3.
  int s0 = tid * 4;
  int sl = s0 >> 4;        // destination "lane" of these slots
  int si = s0 & 15;        // first element index within that lane (0,4,8,12)
  int sg = (sl >> 4) << 4; // A k-group offset (0 or 16)
  int kbase_s = sg + si;  // k-delta of slot j=0 (j adds +1)
  int ms = m0 + (sl & 15);
  int qy_s = ms / Wq, qx_s = ms - qy_s * Wq;
  int iy_base = qy_s * stride - 1;
  int ix_base = qx_s * stride - 1;

  v8f acc0 = {}, acc1 = {};
  for (int kb = 0; kb < K; kb += 64) {
    __builtin_prefetch((const void*)(wl0 + (size_t)((kb >> 5) + 2) * wchunk), 0, 1);
    // stage chunk kb into buf 0
#pragma unroll
    for (int j = 0; j < 4; ++j) {
      int k = kb + kbase_s + j;
      int ci = k / 9;
      int t = k - ci * 9;
      int ry = t / 3;
      int rx = t - ry * 3;
      int iy = iy_base + ry;
      int ix = ix_base + rx;
      _Float16 v = (_Float16)0.f;
      if ((unsigned)iy < (unsigned)H && (unsigned)ix < (unsigned)W)
        v = xb[((size_t)ci * H + iy) * W + ix];
      ldsA[0][s0 + j] = v;
    }
    __syncthreads();
    v16h a0 = *(const v16h*)&ldsA[0][lane << 4];
    v16h b00 = *(const v16h*)(wl0 + (size_t)(kb >> 5) * wchunk);
    v16h b01 = *(const v16h*)(wl1 + (size_t)(kb >> 5) * wchunk);
    // stage chunk kb+32 into buf 1
#pragma unroll
    for (int j = 0; j < 4; ++j) {
      int k = kb + 32 + kbase_s + j;
      int ci = k / 9;
      int t = k - ci * 9;
      int ry = t / 3;
      int rx = t - ry * 3;
      int iy = iy_base + ry;
      int ix = ix_base + rx;
      _Float16 v = (_Float16)0.f;
      if ((unsigned)iy < (unsigned)H && (unsigned)ix < (unsigned)W)
        v = xb[((size_t)ci * H + iy) * W + ix];
      ldsA[1][s0 + j] = v;
    }
    __syncthreads();
    v16h a1 = *(const v16h*)&ldsA[1][lane << 4];
    v16h b10 = *(const v16h*)(wl0 + (size_t)((kb >> 5) + 1) * wchunk);
    v16h b11 = *(const v16h*)(wl1 + (size_t)((kb >> 5) + 1) * wchunk);
    acc0 = __builtin_amdgcn_wmma_f32_16x16x32_f16(false, a0, false, b00, (short)0, acc0,
                                                  false, false);
    acc1 = __builtin_amdgcn_wmma_f32_16x16x32_f16(false, a0, false, b01, (short)0, acc1,
                                                  false, false);
    acc0 = __builtin_amdgcn_wmma_f32_16x16x32_f16(false, a1, false, b10, (short)0, acc0,
                                                  false, false);
    acc1 = __builtin_amdgcn_wmma_f32_16x16x32_f16(false, a1, false, b11, (short)0, acc1,
                                                  false, false);
  }

  // Epilogue: +noise*nw, +bias, leaky_relu, *actScale, store f16
  int co0 = n0 + n;
  int co1 = co0 + 16;
  int rbase = gb << 3;
  float nw = nwp ? nwp[0] : 0.f;
  float bias0 = ab[co0];
  float bias1 = ab[co1];
  size_t HoWo = (size_t)Ho * Wo;
  _Float16* ob0 = out + ((size_t)b * Cout + co0) * HoWo;
  _Float16* ob1 = out + ((size_t)b * Cout + co1) * HoWo;
#pragma unroll
  for (int r = 0; r < 8; ++r) {
    int mm = m0 + rbase + r;
    int oqy = mm / Wq, oqx = mm - oqy * Wq;
    int oy = upMode ? (oqy * 2 + py) : oqy;
    int ox = upMode ? (oqx * 2 + px) : oqx;
    float nz = noise ? nw * noise[oy * Wo + ox] : 0.f;
    float v0 = lrelu(acc0[r] + nz + bias0) * actScale;
    float v1 = lrelu(acc1[r] + nz + bias1) * actScale;
    size_t off = (size_t)oy * Wo + ox;
    ob0[off] = (_Float16)v0;
    ob1[off] = (_Float16)v1;
  }
}

// ---------------------------------------------------------------------------
// to_rgb weight prep: rgbw[b][o][ci] = (1/sqrt(Cin)) * W[o][ci] * style[b][ci]
// ---------------------------------------------------------------------------
__global__ void k_rgbw(const float* __restrict__ W, const float* __restrict__ style,
                       float* __restrict__ rgbw, int Cin, float scale) {
  int idx = blockIdx.x * blockDim.x + threadIdx.x;
  if (idx >= 12 * Cin) return;
  int ci = idx % Cin;
  int o = (idx / Cin) % 3;
  int b = idx / (3 * Cin);
  rgbw[idx] = scale * W[o * Cin + ci] * style[b * Cin + ci];
}

// ---------------------------------------------------------------------------
// to_rgb: 1x1 modulated conv (all 3 channels per pixel-thread) + bias +
// blur-upsampled skip.  f32 out.
// ---------------------------------------------------------------------------
__global__ void k_to_rgb(const _Float16* __restrict__ x, const float* __restrict__ rgbw,
                         const float* __restrict__ bias, const float* __restrict__ skip,
                         float* __restrict__ out, int Cin, int H, int Hs) {
  int HW = H * H;
  int idx = blockIdx.x * blockDim.x + threadIdx.x;
  if (idx >= 4 * HW) return;
  int b = idx / HW;
  int p = idx - b * HW;
  int oy = p / H, ox = p - oy * H;
  const _Float16* xb = x + (size_t)b * Cin * HW + p;
  const float* w0 = rgbw + (size_t)b * 3 * Cin;
  const float* w1 = w0 + Cin;
  const float* w2 = w1 + Cin;
  float a0 = 0.f, a1 = 0.f, a2 = 0.f;
  for (int ci = 0; ci < Cin; ++ci) {
    float xv = (float)xb[(size_t)ci * HW];
    a0 += xv * w0[ci];
    a1 += xv * w1[ci];
    a2 += xv * w2[ci];
  }
  float acc[3] = {a0 + bias[0], a1 + bias[1], a2 + bias[2]};
  if (skip) {
    int iy0, ix0;
    float wy[2], wx[2];
    if (oy & 1) { iy0 = (oy - 1) >> 1; wy[0] = 3.f; wy[1] = 1.f; }
    else        { iy0 = (oy >> 1) - 1; wy[0] = 1.f; wy[1] = 3.f; }
    if (ox & 1) { ix0 = (ox - 1) >> 1; wx[0] = 3.f; wx[1] = 1.f; }
    else        { ix0 = (ox >> 1) - 1; wx[0] = 1.f; wx[1] = 3.f; }
    for (int o = 0; o < 3; ++o) {
      const float* sp = skip + ((size_t)b * 3 + o) * Hs * Hs;
      float s = 0.f;
      for (int a = 0; a < 2; ++a) {
        int iy = iy0 + a;
        if ((unsigned)iy >= (unsigned)Hs) continue;
        for (int c2 = 0; c2 < 2; ++c2) {
          int ix = ix0 + c2;
          if ((unsigned)ix >= (unsigned)Hs) continue;
          s += wy[a] * wx[c2] * sp[iy * Hs + ix];
        }
      }
      acc[o] += s * (1.f / 16.f);
    }
  }
  for (int o = 0; o < 3; ++o) out[((size_t)b * 3 + o) * HW + p] = acc[o];
}

// ---------------------------------------------------------------------------
// cond_map: linear (no scale), conv1 (Cin=1, stride2), conv2 via WMMA path
// ---------------------------------------------------------------------------
__global__ void k_cond_linear(const float* __restrict__ cond, const float* __restrict__ lw,
                              const float* __restrict__ lb, float* __restrict__ h0) {
  int idx = blockIdx.x * blockDim.x + threadIdx.x;  // 4*256
  if (idx >= 1024) return;
  int b = idx >> 8, o = idx & 255;
  float acc = 0.f;
  const float* cr = cond + b * 512;
  const float* wr = lw + (size_t)o * 512;
  for (int i = 0; i < 512; ++i) acc += cr[i] * wr[i];
  h0[idx] = acc + lb[o];
}

__global__ void k_cond_conv1(const float* __restrict__ h0, const float* __restrict__ w,
                             const float* __restrict__ bias, _Float16* __restrict__ out) {
  int idx = blockIdx.x * blockDim.x + threadIdx.x;  // 4*512*64
  if (idx >= 4 * 512 * 64) return;
  int p = idx & 63;
  int co = (idx >> 6) & 511;
  int b = idx >> 15;
  int oy = p >> 3, ox = p & 7;
  float acc = 0.f;
  const float* hb = h0 + b * 256;
  for (int ky = 0; ky < 3; ++ky) {
    int iy = 2 * oy - 1 + ky;
    if ((unsigned)iy >= 16u) continue;
    for (int kx = 0; kx < 3; ++kx) {
      int ix = 2 * ox - 1 + kx;
      if ((unsigned)ix >= 16u) continue;
      acc += hb[iy * 16 + ix] * w[co * 9 + ky * 3 + kx];
    }
  }
  acc = lrelu(acc + bias[co]);
  out[idx] = (_Float16)acc;
}

// ---------------------------------------------------------------------------
// Host side
// ---------------------------------------------------------------------------
struct ConvL { int cin, cout, up; };
static const ConvL g_convs[13] = {
    {512, 512, 0}, {512, 512, 1}, {512, 512, 0}, {512, 512, 1}, {512, 512, 0},
    {512, 512, 1}, {512, 512, 0}, {512, 512, 1}, {512, 512, 0}, {512, 256, 1},
    {256, 256, 0}, {256, 128, 1}, {128, 128, 0}};
static const int g_rgb_cin[7] = {512, 512, 512, 512, 512, 256, 128};

extern "C" void kernel_launch(void* const* d_in, const int* in_sizes, int n_in,
                              void* d_out, int out_size, void* d_ws, size_t ws_size,
                              hipStream_t stream) {
  (void)in_sizes; (void)n_in; (void)out_size; (void)ws_size;
  const float* styles = (const float*)d_in[0];
  const float* condition = (const float*)d_in[1];
  const float* mapW[4], *mapB[4];
  for (int i = 0; i < 4; ++i) { mapW[i] = (const float*)d_in[2 + 2 * i]; mapB[i] = (const float*)d_in[3 + 2 * i]; }
  const float* lw  = (const float*)d_in[10];
  const float* lb  = (const float*)d_in[11];
  const float* c1w = (const float*)d_in[12];
  const float* c1b = (const float*)d_in[13];
  const float* c2w = (const float*)d_in[14];
  const float* c2b = (const float*)d_in[15];

  // Workspace carve-up
  char* wsp = (char*)d_ws;
  auto alloc = [&](size_t bytes) -> void* {
    void* p = (void*)wsp;
    wsp += (bytes + 255) & ~(size_t)255;
    return p;
  };
  float*     z      = (float*)alloc(4 * 512 * sizeof(float));
  float*     styleb = (float*)alloc(4 * 512 * sizeof(float));
  float*     demodb = (float*)alloc(4 * 512 * sizeof(float));
  float*     rgbwb  = (float*)alloc(4 * 3 * 512 * sizeof(float));
  float*     h0     = (float*)alloc(4 * 256 * sizeof(float));
  _Float16*  actA   = (_Float16*)alloc((size_t)4 * 512 * 64 * 64 * 2);
  _Float16*  actB   = (_Float16*)alloc((size_t)4 * 512 * 64 * 64 * 2);
  float*     skipA  = (float*)alloc((size_t)4 * 3 * 256 * 256 * 4);
  float*     skipB  = (float*)alloc((size_t)4 * 3 * 256 * 256 * 4);
  _Float16*  wbuf   = (_Float16*)alloc((size_t)16 * 4608 * 512 * 2);

  hipStream_t s = stream;

  // ---- mapping ----
  k_mapping<<<1, 512, 0, s>>>(styles, mapW[0], mapB[0], mapW[1], mapB[1], mapW[2],
                              mapB[2], mapW[3], mapB[3], z);

  // ---- cond_map ----
  k_cond_linear<<<4, 256, 0, s>>>(condition, lw, lb, h0);
  k_cond_conv1<<<(4 * 512 * 64 + 255) / 256, 256, 0, s>>>(h0, c1w, c1b, actA);
  {
    long long tot = 512LL * 512 * 9;
    k_prep_plainw<<<(unsigned)((tot + 255) / 256), 256, 0, s>>>(c2w, wbuf, 512, 512);
    dim3 grid(1, 512 / 128, 4), block(32, 4, 1);
    k_conv_wmma<<<grid, block, 0, s>>>(actA, wbuf, (size_t)0, actB, nullptr, nullptr,
                                       c2b, 512, 512, 8, 8, 4, 4, 2, 0, 1.0f);
  }

  // ---- styled conv launcher ----
  auto styled = [&](int li, const _Float16* xin, _Float16* xout, int H, int Ho, int up) {
    const float* W   = (const float*)d_in[16 + 5 * li + 0];
    const float* sW  = (const float*)d_in[16 + 5 * li + 1];
    const float* sb  = (const float*)d_in[16 + 5 * li + 2];
    const float* nwp = (const float*)d_in[16 + 5 * li + 3];
    const float* ab  = (const float*)d_in[16 + 5 * li + 4];
    const float* noi = (const float*)d_in[109 + li];
    int Cin = g_convs[li].cin, Cout = g_convs[li].cout;
    float mscale = 1.0f / sqrtf((float)(Cin * 9));
    k_style<<<(4 * Cin + 255) / 256, 256, 0, s>>>(z, sW, sb, styleb, Cin);
    k_demod<<<4 * Cout, 256, 0, s>>>(W, styleb, demodb, Cin, Cout, mscale);
    if (up) {
      long long tot = 4LL * Cout * Cin;
      k_prepw_up<<<(unsigned)((tot + 255) / 256), 256, 0, s>>>(W, styleb, demodb, wbuf,
                                                              Cin, Cout, mscale);
    } else {
      long long tot = 4LL * Cin * 9 * Cout;
      k_prepw<<<(unsigned)((tot + 255) / 256), 256, 0, s>>>(W, styleb, demodb, wbuf,
                                                           Cin, Cout, mscale);
    }
    size_t wz = (size_t)Cin * 9 * Cout;
    int Hq = up ? Ho / 2 : Ho;
    dim3 grid((Hq * Hq) / 16, Cout / 128, up ? 16 : 4), block(32, 4, 1);
    k_conv_wmma<<<grid, block, 0, s>>>(xin, wbuf, wz, xout, noi, nwp, ab, Cin, Cout, H,
                                       H, Ho, Ho, 1, up, SQRT2F);
  };

  auto rgb = [&](int ri, const _Float16* x, int H, const float* skip, int Hs,
                 float* outp) {
    const float* W  = (const float*)d_in[81 + 4 * ri + 0];
    const float* sW = (const float*)d_in[81 + 4 * ri + 1];
    const float* sb = (const float*)d_in[81 + 4 * ri + 2];
    const float* bi = (const float*)d_in[81 + 4 * ri + 3];
    int Cin = g_rgb_cin[ri];
    k_style<<<(4 * Cin + 255) / 256, 256, 0, s>>>(z, sW, sb, styleb, Cin);
    k_rgbw<<<(12 * Cin + 255) / 256, 256, 0, s>>>(W, styleb, rgbwb, Cin,
                                                  1.0f / sqrtf((float)Cin));
    k_to_rgb<<<(4 * H * H + 255) / 256, 256, 0, s>>>(x, rgbwb, bi, skip, outp, Cin, H,
                                                     Hs);
  };

  // ---- main synthesis ----
  styled(0, actB, actA, 4, 4, 0);          // conv0 @4x4   (actB -> actA)
  rgb(0, actA, 4, nullptr, 0, skipA);      // skip @4x4
  float* skin = skipA;
  float* skout = skipB;
  for (int j = 0; j < 6; ++j) {
    int Hin = 4 << j, Hout = 8 << j;
    styled(1 + 2 * j, actA, actB, Hin, Hout, 1);   // up conv   (actA -> actB)
    styled(2 + 2 * j, actB, actA, Hout, Hout, 0);  // conv      (actB -> actA)
    float* dst = (j == 5) ? (float*)d_out : skout;
    rgb(1 + j, actA, Hout, skin, Hin, dst);
    float* tmp = skin; skin = skout; skout = tmp;
  }
}
